// Model_32993938768243
// MI455X (gfx1250) — matrix operand order repaired and checked by reading, not compiled, not run on hardware
//
#include <hip/hip_runtime.h>
#include <hip/hip_bf16.h>

// Problem constants (match reference)
#define B_    512
#define T_    256
#define H_    512
#define DIN_  514
#define DOUT_ 2
#define KPAD_ 544    // DIN padded up to a multiple of 32 (17*32)
#define DTAU_ 0.1f

typedef __attribute__((ext_vector_type(16))) __bf16 bf16x16;
typedef __attribute__((ext_vector_type(8)))  __bf16 bf16x8;
typedef __attribute__((ext_vector_type(8)))  float  f32x8;
typedef __attribute__((ext_vector_type(2)))  float  f32x2;

// ---------------------------------------------------------------------------
// A-fragment loader (16x32 bf16, row-major source in LDS).
// Layout: lanes 0-15 are rows M=0..15 with k = kh*16+0..15 (contiguous),
// kh = lane>>4. Two contiguous 16B LDS loads per lane.
// ---------------------------------------------------------------------------
__device__ __forceinline__ bf16x16 load_a_frag_lds(const __bf16* base, int stride,
                                                   int kbase, int row, int kh) {
  const __bf16* p = base + row * stride + kbase + kh * 16;
  bf16x8 lo = *(const bf16x8*)(p);
  bf16x8 hi = *(const bf16x8*)(p + 8);
  return __builtin_shufflevector(lo, hi, 0,1,2,3,4,5,6,7,8,9,10,11,12,13,14,15);
}

// ---------------------------------------------------------------------------
// B-fragment loader (32x16 bf16): lane holds 16 contiguous k-values of one
// weight row (= output column): one 32-byte contiguous load (2x b128).
// Plain loads (default scope, L2 line-shared by all waves, clause-able);
// anti-hoisting is handled by laundering the base pointer per timestep.
// ---------------------------------------------------------------------------
__device__ __forceinline__ bf16x16 load_b_frag(const __bf16* p) {
  bf16x8 lo = *(const bf16x8*)(p);
  bf16x8 hi = *(const bf16x8*)(p + 8);
  return __builtin_shufflevector(lo, hi, 0,1,2,3,4,5,6,7,8,9,10,11,12,13,14,15);
}

// Load the 4 N-tile B-fragments of one k-chunk.
__device__ __forceinline__ void load_b4(bf16x16 dst[4], const __bf16* base,
                                        const int off[4], int kc) {
#pragma unroll
  for (int nt = 0; nt < 4; ++nt) dst[nt] = load_b_frag(base + off[nt] + kc * 32);
}

// 4 WMMAs of one k-chunk (shared A-fragment).
__device__ __forceinline__ void mma4(f32x8 acc[4], const bf16x16 frag[4], bf16x16 a) {
#pragma unroll
  for (int nt = 0; nt < 4; ++nt)
    acc[nt] = __builtin_amdgcn_wmma_f32_16x16x32_bf16(
        false, a, false, frag[nt], (short)0, acc[nt], false, false);
}

// ---------------------------------------------------------------------------
// Weight conversion: fp32 -> bf16 (once; both copies stay L2-resident)
// ---------------------------------------------------------------------------
__global__ void convert_wh_kernel(const float* __restrict__ W, __bf16* __restrict__ Wb) {
  int idx = blockIdx.x * 256 + threadIdx.x;
  if (idx < H_ * H_) Wb[idx] = (__bf16)W[idx];
}

__global__ void convert_wx_kernel(const float* __restrict__ W, __bf16* __restrict__ Wb) {
  int idx = blockIdx.x * 256 + threadIdx.x;
  if (idx < H_ * KPAD_) {
    int n = idx / KPAD_;
    int k = idx - n * KPAD_;
    Wb[idx] = (k < DIN_) ? (__bf16)W[n * DIN_ + k] : (__bf16)0.f;
  }
}

// ---------------------------------------------------------------------------
// Fused recurrent kernel.
// Grid: 32 workgroups, each owns a 16-row batch tile for all T steps (the
// scan is only parallel over batch; h[b,:] never crosses tiles -> no inter-WG
// sync). Block: 256 threads = 8 waves; wave w owns output columns
// [w*64, w*64+64) = 4 N-tiles. Per step per wave: (17+16) k-chunks x 4
// N-tiles bf16 WMMA with fp32 accumulation; ah state lives in fp32 VGPRs.
// Ping-pong B-fragment buffers (no copy chain); noise preloaded before the
// GEMMs; hstore writes deferred one step so STOREcnt drains behind WMMAs.
// ---------------------------------------------------------------------------
__global__ __launch_bounds__(256) void rnn_scan_kernel(
    const float*  __restrict__ x,      // [B,T,DIN]
    const float*  __restrict__ noise,  // [B,T,H]
    const float*  __restrict__ b_ah,   // [H]
    const float*  __restrict__ ah0,    // [H]
    const __bf16* __restrict__ Whb,    // [H,H]    bf16
    const __bf16* __restrict__ Wxb,    // [H,KPAD] bf16, zero-padded
    float*        __restrict__ hstore) // [B,T,H]
{
  __shared__ __bf16 lds_x[16 * KPAD_]; // x_t tile (bf16, padded)   17 KB
  __shared__ __bf16 lds_h[16 * H_];    // current h tile (bf16)     16 KB

  const int tid  = threadIdx.x;
  const int lane = tid & 31;
  const int wave = tid >> 5;
  const int nl   = lane & 15;  // A-frag row / B,C-frag column
  const int kh   = lane >> 4;  // k-half (A/B) and m-half (C) selector
  const int mb   = kh * 8;     // C-frag row base
  const int b0   = blockIdx.x * 16;

  int   ng[4];
  float biasv[4];
  f32x8 ah[4];
  int   offx[4];          // per-N-tile element offset of Wx B-fragments
  int   offh[4];          // per-N-tile element offset of Wh B-fragments
  float hsav[4][8];       // previous step's h values, pending global store
#pragma unroll
  for (int nt = 0; nt < 4; ++nt) {
    ng[nt]    = wave * 64 + nt * 16 + nl;
    biasv[nt] = b_ah[ng[nt]];
    float a0  = ah0[ng[nt]];
    ah[nt]    = (f32x8){a0, a0, a0, a0, a0, a0, a0, a0};
    offx[nt]  = ng[nt] * KPAD_ + kh * 16;
    offh[nt]  = ng[nt] * H_   + kh * 16;
  }

  // h_init = retanh(ah0), broadcast over the 16 batch rows
  for (int idx = tid; idx < 16 * H_; idx += 256) {
    int n = idx & (H_ - 1);
    lds_h[idx] = (__bf16)fmaxf(tanhf(ah0[n]), 0.f);
  }
  __syncthreads();

  for (int t = 0; t < T_; ++t) {
    // --- stage x[b0..b0+15, t, :] into LDS as bf16 (pairwise; pad is whole
    // pairs since DIN is even: pairs 0..257 read from x (257 lies past DIN), 258..271 zero) ----------
    for (int r = 0; r < 16; ++r) {
      const f32x2* xr2 = (const f32x2*)(x + ((size_t)(b0 + r) * T_ + t) * DIN_);
      for (int c2 = tid; c2 < KPAD_ / 2; c2 += 256) {
        f32x2 v = (c2 < (DIN_ / 2 + 1)) ? xr2[c2] : (f32x2){0.f, 0.f};
        lds_x[r * KPAD_ + 2 * c2]     = (__bf16)v.x;
        lds_x[r * KPAD_ + 2 * c2 + 1] = (__bf16)v.y;
      }
    }
    // prefetch next step's x slab (global_prefetch_b8); 16 rows x 9 lines
    if (t + 1 < T_) {
      for (int p = tid; p < 16 * 9; p += 256) {
        int r = p & 15, j = p >> 4;
        __builtin_prefetch(x + ((size_t)(b0 + r) * T_ + (t + 1)) * DIN_ + j * 64, 0, 3);
      }
    }
    __syncthreads();

    // Launder the (uniform) weight base pointers through an opaque asm once
    // per timestep so LLVM cannot prove the B-fragment loads t-invariant
    // (round-1: hoist + scratch spill). Loads stay ordinary global_loads.
    const __bf16* wxv = Wxb;
    const __bf16* whv = Whb;
    asm volatile("" : "+s"(wxv), "+s"(whv));

    // Deferred hstore writes from step t-1: issued here so the store-acks
    // drain behind this step's GEMMs instead of stalling at the barrier.
    if (t > 0) {
#pragma unroll
      for (int nt = 0; nt < 4; ++nt)
#pragma unroll
        for (int i = 0; i < 8; ++i)
          hstore[((size_t)(b0 + mb + i) * T_ + (t - 1)) * H_ + ng[nt]] = hsav[nt][i];
    }
    // Preload this step's noise: independent of the GEMMs, so its HBM
    // latency hides behind 132 WMMAs.
    float nz[4][8];
#pragma unroll
    for (int nt = 0; nt < 4; ++nt)
#pragma unroll
      for (int i = 0; i < 8; ++i)
        nz[nt][i] = noise[((size_t)(b0 + mb + i) * T_ + t) * H_ + ng[nt]];

    f32x8 acc[4];
#pragma unroll
    for (int nt = 0; nt < 4; ++nt) acc[nt] = (f32x8){0, 0, 0, 0, 0, 0, 0, 0};

    bf16x16 fA[4], fB[4];

    // --- drive: x_t @ W_x_ah^T (17 chunks), ping-pong fragments -----------
    load_b4(fA, wxv, offx, 0);
    for (int kc = 0; kc < 16; kc += 2) {
      load_b4(fB, wxv, offx, kc + 1);
      mma4(acc, fA, load_a_frag_lds(lds_x, KPAD_, kc * 32, nl, kh));
      if (kc + 2 < 17) load_b4(fA, wxv, offx, kc + 2);
      mma4(acc, fB, load_a_frag_lds(lds_x, KPAD_, (kc + 1) * 32, nl, kh));
    }
    // leftover drive chunk 16 (in fA); preload first recurrent chunk
    load_b4(fB, whv, offh, 0);
    mma4(acc, fA, load_a_frag_lds(lds_x, KPAD_, 16 * 32, nl, kh));

    // --- recurrence: h @ W_h_ah^T (16 chunks), ping-pong fragments --------
    for (int kc = 0; kc < 16; kc += 2) {
      if (kc + 1 < 16) load_b4(fA, whv, offh, kc + 1);
      mma4(acc, fB, load_a_frag_lds(lds_h, H_, kc * 32, nl, kh));
      if (kc + 2 < 16) load_b4(fB, whv, offh, kc + 2);
      mma4(acc, fA, load_a_frag_lds(lds_h, H_, (kc + 1) * 32, nl, kh));
    }
    __syncthreads();  // all waves done reading lds_h/lds_x before rewrite

    // --- state update: ah fp32 in regs, h -> LDS (bf16) + pending store ---
#pragma unroll
    for (int nt = 0; nt < 4; ++nt) {
#pragma unroll
      for (int i = 0; i < 8; ++i) {
        float av = ah[nt][i];
        float nv = av + DTAU_ * (acc[nt][i] + biasv[nt] - av);
        float hn = fmaxf(tanhf(nv), 0.f) + nz[nt][i];
        hsav[nt][i] = hn;
        lds_h[(mb + i) * H_ + ng[nt]] = (__bf16)hn;
        ah[nt][i] = nv;
      }
    }
    // next iteration's post-x-stage __syncthreads publishes lds_h updates
  }

  // flush the final step's h values
#pragma unroll
  for (int nt = 0; nt < 4; ++nt)
#pragma unroll
    for (int i = 0; i < 8; ++i)
      hstore[((size_t)(b0 + mb + i) * T_ + (T_ - 1)) * H_ + ng[nt]] = hsav[nt][i];
}

// ---------------------------------------------------------------------------
// Output projection: out[b,t,o] = dot(hstore[b,t,:], W_h_y[o,:]), DOUT = 2.
// One wave per (b,t) row; coalesced column-strided loads + shuffle reduction.
// ---------------------------------------------------------------------------
__global__ __launch_bounds__(256) void out_proj_kernel(
    const float* __restrict__ hstore, const float* __restrict__ Wy,
    float* __restrict__ out)
{
  const int lane = threadIdx.x & 31;
  const int wave = threadIdx.x >> 5;
  const int row  = blockIdx.x * 8 + wave;  // flat (b*T + t)
  const float* h = hstore + (size_t)row * H_;
  float s0 = 0.f, s1 = 0.f;
  for (int c = lane; c < H_; c += 32) {
    float hv = h[c];
    s0 += hv * Wy[c];
    s1 += hv * Wy[H_ + c];
  }
  for (int off = 16; off > 0; off >>= 1) {
    s0 += __shfl_down(s0, off);
    s1 += __shfl_down(s1, off);
  }
  if (lane == 0) {
    out[(size_t)row * DOUT_ + 0] = s0;
    out[(size_t)row * DOUT_ + 1] = s1;
  }
}

// ---------------------------------------------------------------------------
extern "C" void kernel_launch(void* const* d_in, const int* in_sizes, int n_in,
                              void* d_out, int out_size, void* d_ws, size_t ws_size,
                              hipStream_t stream) {
  (void)in_sizes; (void)n_in; (void)out_size; (void)ws_size;

  const float* x     = (const float*)d_in[0];  // [B,T,DIN]
  const float* noise = (const float*)d_in[1];  // [B,T,H]
  const float* Wx    = (const float*)d_in[2];  // [H,DIN]
  const float* bah   = (const float*)d_in[3];  // [H]
  const float* Wh    = (const float*)d_in[4];  // [H,H]
  const float* Wy    = (const float*)d_in[5];  // [DOUT,H]
  const float* ah0   = (const float*)d_in[6];  // [H]

  float* out    = (float*)d_out;                       // [B,T,DOUT]
  float* hstore = out + (size_t)B_ * T_ * DOUT_;       // [B,T,H]

  // workspace: bf16 weight copies (~1.05 MB total, stays in L2)
  char*   ws  = (char*)d_ws;
  __bf16* Whb = (__bf16*)ws;
  __bf16* Wxb = (__bf16*)(ws + (size_t)H_ * H_ * sizeof(__bf16));

  hipLaunchKernelGGL(convert_wh_kernel, dim3((H_ * H_ + 255) / 256), dim3(256), 0, stream,
                     Wh, Whb);
  hipLaunchKernelGGL(convert_wx_kernel, dim3((H_ * KPAD_ + 255) / 256), dim3(256), 0, stream,
                     Wx, Wxb);
  hipLaunchKernelGGL(rnn_scan_kernel, dim3(B_ / 16), dim3(256), 0, stream,
                     x, noise, bah, ah0, Whb, Wxb, hstore);
  hipLaunchKernelGGL(out_proj_kernel, dim3(B_ * T_ / 8), dim3(256), 0, stream,
                     hstore, Wy, out);
}
